// SceneMultimodalCrossAttnTransformerLayer_34522947125327
// MI455X (gfx1250) — compile-verified, matrix operand order checked by reading, not run on hardware
//
#include <hip/hip_runtime.h>
#include <hip/hip_bf16.h>

typedef __attribute__((ext_vector_type(16))) _Float16 v16h;
typedef __attribute__((ext_vector_type(8)))  _Float16 v8h;
typedef __attribute__((ext_vector_type(8)))  float    v8f;
typedef __attribute__((ext_vector_type(4)))  float    v4f;

#define Hdim 256
#define FDIM 128
#define NTOK (16*32*64*8)      /* 262144 tokens */
#define LN_EPSF 1e-6f
#define WPB 4                  /* waves per block */
#define X1S 264                /* x1 LDS row stride in halfs (padded) */
#define FFS_S 136              /* ff LDS row stride in halfs (padded) */

// Build a 16-half WMMA fragment (A or B operand) from two contiguous 8-half
// chunks at p and p+16 halfs (per the gfx1250 16-bit operand VGPR layout:
// halves 0..7 = K base..base+7, halves 8..15 = K base+16..base+23).
__device__ __forceinline__ v16h frag16(const _Float16* p) {
  v8h lo = *(const v8h*)(p);
  v8h hi = *(const v8h*)(p + 16);
  v16h a;
#pragma unroll
  for (int i = 0; i < 8; ++i) { a[i] = lo[i]; a[i + 8] = hi[i]; }
  return a;
}

// ---------------------------------------------------------------------------
// Prologue 1: per-batch constant attention output.
// softmax over S=1 is identically 1, so attn_out row = (scene@Wv + bv)@Wo + bo.
// One block per batch (16 blocks, 256 threads).
__global__ void base_kernel(const float* __restrict__ scene,
                            const float* __restrict__ Wv, const float* __restrict__ bv,
                            const float* __restrict__ Wo, const float* __restrict__ bo,
                            float* __restrict__ basev) {
  __shared__ float sc[Hdim];
  __shared__ float vv[Hdim];
  const int b = blockIdx.x, t = threadIdx.x;
  sc[t] = scene[b * Hdim + t];
  __syncthreads();
  float a = bv[t];
#pragma unroll 8
  for (int h = 0; h < Hdim; ++h) a += sc[h] * Wv[h * Hdim + t];   // Wv [H][E*D]
  vv[t] = a;
  __syncthreads();
  float o = bo[t];
#pragma unroll 8
  for (int ed = 0; ed < Hdim; ++ed) o += vv[ed] * Wo[ed * Hdim + t]; // Wo [E*D][H]
  basev[b * Hdim + t] = o;
}

// Prologue 2: f32 -> f16 transposed weights so B-fragments are contiguous
// 16B global loads. w1t = W1^T [128][256], w2t = W2^T [256][128].
__global__ void cvt_kernel(const float* __restrict__ W1, const float* __restrict__ W2,
                           _Float16* __restrict__ w1t, _Float16* __restrict__ w2t) {
  const int idx = blockIdx.x * 256 + threadIdx.x;    // 0..32767
  const int n1 = idx >> 8, k1 = idx & 255;
  w1t[idx] = (_Float16)W1[k1 * FDIM + n1];
  const int n2 = idx >> 7, k2 = idx & 127;
  w2t[idx] = (_Float16)W2[k2 * Hdim + n2];
}

// Prologue 3: attn_score == softmax over size-1 axis == all ones.
__global__ void fill_ones(float* __restrict__ p, int n) {
  const int i = blockIdx.x * 256 + threadIdx.x;
  if (i < n) p[i] = 1.0f;
}

// ---------------------------------------------------------------------------
// Fused main kernel: x1 = LN(hidden + base)*g1+b1ln; ff = relu(x1@W1+b1);
// y = ff@W2+b2; out = LN(y + x1)*g2+b2ln.  One 16-token tile per wave.
__global__ __launch_bounds__(WPB * 32)
void layer_kernel(const float* __restrict__ hidden, const float* __restrict__ base,
                  const _Float16* __restrict__ w1t, const _Float16* __restrict__ w2t,
                  const float* __restrict__ b1, const float* __restrict__ b2,
                  const float* __restrict__ g1, const float* __restrict__ o1,
                  const float* __restrict__ g2, const float* __restrict__ o2,
                  float* __restrict__ out) {
  __shared__ _Float16 x1s[WPB][16 * X1S];   // 16x256 f16 per wave, padded
  __shared__ _Float16 ffs[WPB][16 * FFS_S]; // 16x128 f16 per wave, padded

  const int lane = threadIdx.x & 31;
  const int wave = threadIdx.x >> 5;
  const int tile = blockIdx.x * WPB + wave;
  const long row0 = (long)tile * 16;
  const int bidx = (int)(row0 >> 14);       // token / (A*T*N) ; tiles never span batches

  const int c0 = lane * 8;
  const float* __restrict__ hid = hidden + row0 * Hdim;
  __builtin_prefetch(hid, 0, 1);            // global_prefetch_b8

  const v4f base0 = *(const v4f*)(base + bidx * Hdim + c0);
  const v4f base1 = *(const v4f*)(base + bidx * Hdim + c0 + 4);
  const v4f g1a = *(const v4f*)(g1 + c0), g1b = *(const v4f*)(g1 + c0 + 4);
  const v4f o1a = *(const v4f*)(o1 + c0), o1b = *(const v4f*)(o1 + c0 + 4);

  // ---- Phase 1: LayerNorm1 -> x1 (f16, LDS) ----
#pragma unroll 4
  for (int m = 0; m < 16; ++m) {
    const float* hr = hid + m * Hdim + c0;
    const v4f a = *(const v4f*)(hr);
    const v4f b = *(const v4f*)(hr + 4);
    float v[8];
#pragma unroll
    for (int i = 0; i < 4; ++i) { v[i] = a[i] + base0[i]; v[4 + i] = b[i] + base1[i]; }
    float s = 0.f, q = 0.f;
#pragma unroll
    for (int i = 0; i < 8; ++i) { s += v[i]; q += v[i] * v[i]; }
#pragma unroll
    for (int off = 1; off < 32; off <<= 1) {
      s += __shfl_xor(s, off, 32);
      q += __shfl_xor(q, off, 32);
    }
    const float mu = s * (1.f / Hdim);
    const float var = q * (1.f / Hdim) - mu * mu;
    const float rs = rsqrtf(var + LN_EPSF);
    v8h xh;
#pragma unroll
    for (int i = 0; i < 4; ++i) {
      xh[i]     = (_Float16)((v[i]     - mu) * rs * g1a[i] + o1a[i]);
      xh[4 + i] = (_Float16)((v[4 + i] - mu) * rs * g1b[i] + o1b[i]);
    }
    *(v8h*)&x1s[wave][m * X1S + c0] = xh;   // ds_write_b128, 16B aligned
  }
  __syncthreads();

  const int nloc = lane & 15;
  const int khi  = (lane < 16) ? 0 : 8;     // K sub-offset per lane half
  const int mhi  = (lane >> 4) * 8;         // C/D row offset per lane half

  // ---- Phase 2: GEMM1  [16x256]x[256x128] via v_wmma_f32_16x16x32_f16 ----
  v8f acc1[8];
#pragma unroll
  for (int nt = 0; nt < 8; ++nt)
#pragma unroll
    for (int i = 0; i < 8; ++i) acc1[nt][i] = 0.f;

#pragma unroll
  for (int kk = 0; kk < 8; ++kk) {
    const v16h afrag = frag16(&x1s[wave][(lane & 15) * X1S + kk * 32 + khi]);
#pragma unroll
    for (int nt = 0; nt < 8; ++nt) {
      const v16h bfrag = frag16(w1t + (size_t)(nt * 16 + nloc) * Hdim + kk * 32 + khi);
      acc1[nt] = __builtin_amdgcn_wmma_f32_16x16x32_f16(
          false, afrag, false, bfrag, (short)0, acc1[nt], false, false);
    }
  }
  // bias + relu -> f16 LDS stage (A operand of GEMM2)
#pragma unroll
  for (int nt = 0; nt < 8; ++nt) {
    const float bia = b1[nt * 16 + nloc];
#pragma unroll
    for (int r = 0; r < 8; ++r) {
      const float f = acc1[nt][r] + bia;
      ffs[wave][(mhi + r) * FFS_S + nt * 16 + nloc] = (_Float16)fmaxf(f, 0.f);
    }
  }
  __syncthreads();

  // ---- Phase 3: GEMM2  [16x128]x[128x256] ----
  v8f acc2[16];
#pragma unroll
  for (int nt = 0; nt < 16; ++nt)
#pragma unroll
    for (int i = 0; i < 8; ++i) acc2[nt][i] = 0.f;

#pragma unroll
  for (int kk = 0; kk < 4; ++kk) {
    const v16h afrag = frag16(&ffs[wave][(lane & 15) * FFS_S + kk * 32 + khi]);
#pragma unroll
    for (int nt = 0; nt < 16; ++nt) {
      const v16h bfrag = frag16(w2t + (size_t)(nt * 16 + nloc) * FDIM + kk * 32 + khi);
      acc2[nt] = __builtin_amdgcn_wmma_f32_16x16x32_f16(
          false, afrag, false, bfrag, (short)0, acc2[nt], false, false);
    }
  }

  // ---- Phase 4: residual + LayerNorm2 + store ----
  float bz[16], gz[16], oz[16];
#pragma unroll
  for (int nt = 0; nt < 16; ++nt) {
    const int h = nt * 16 + nloc;
    bz[nt] = b2[h]; gz[nt] = g2[h]; oz[nt] = o2[h];
  }
#pragma unroll
  for (int r = 0; r < 8; ++r) {
    float s = 0.f, q = 0.f;
#pragma unroll
    for (int nt = 0; nt < 16; ++nt) {
      const float x1v = (float)x1s[wave][(mhi + r) * X1S + nt * 16 + nloc];
      const float z = acc2[nt][r] + bz[nt] + x1v;
      acc2[nt][r] = z;
      s += z; q += z * z;
    }
    // rows live in 16-lane half-groups: xor masks 1/2/4/8 reduce within group
#pragma unroll
    for (int off = 1; off < 16; off <<= 1) {
      s += __shfl_xor(s, off, 32);
      q += __shfl_xor(q, off, 32);
    }
    const float mu = s * (1.f / Hdim);
    const float var = q * (1.f / Hdim) - mu * mu;
    const float rs = rsqrtf(var + LN_EPSF);
    float* orow = out + (row0 + mhi + r) * Hdim;
#pragma unroll
    for (int nt = 0; nt < 16; ++nt) {
      orow[nt * 16 + nloc] = (acc2[nt][r] - mu) * rs * gz[nt] + oz[nt];
    }
  }
}

// ---------------------------------------------------------------------------
extern "C" void kernel_launch(void* const* d_in, const int* in_sizes, int n_in,
                              void* d_out, int out_size, void* d_ws, size_t ws_size,
                              hipStream_t stream) {
  (void)in_sizes; (void)n_in; (void)out_size; (void)ws_size;
  const float* hidden = (const float*)d_in[0];
  const float* scene  = (const float*)d_in[1];
  // d_in[2..5] = Wq,bq,Wk,bk are dead (softmax over size-1 axis == 1)
  const float* Wv = (const float*)d_in[6];
  const float* bv = (const float*)d_in[7];
  const float* Wo = (const float*)d_in[8];
  const float* bo = (const float*)d_in[9];
  const float* g1 = (const float*)d_in[10];
  const float* o1 = (const float*)d_in[11];
  const float* W1 = (const float*)d_in[12];
  const float* b1 = (const float*)d_in[13];
  const float* W2 = (const float*)d_in[14];
  const float* b2 = (const float*)d_in[15];
  const float* g2 = (const float*)d_in[16];
  const float* o2 = (const float*)d_in[17];

  float*     basev = (float*)d_ws;                                   // 16*256 f32
  _Float16*  w1t   = (_Float16*)((char*)d_ws + 16 * Hdim * sizeof(float));
  _Float16*  w2t   = w1t + Hdim * FDIM;                              // +32768 halfs

  float* outv  = (float*)d_out;
  float* score = outv + (size_t)NTOK * Hdim;   // attn_score region (all ones)

  base_kernel<<<16, 256, 0, stream>>>(scene, Wv, bv, Wo, bo, basev);
  cvt_kernel<<<128, 256, 0, stream>>>(W1, W2, w1t, w2t);
  fill_ones<<<(NTOK * 8) / 256, 256, 0, stream>>>(score, NTOK * 8);
  layer_kernel<<<NTOK / (16 * WPB), WPB * 32, 0, stream>>>(
      hidden, basev, w1t, w2t, b1, b2, g1, o1, g2, o2, outv);
}
